// WindowProcessor_45449343926782
// MI455X (gfx1250) — compile-verified, matrix operand order checked by reading, not run on hardware
//
#include <hip/hip_runtime.h>
#include <hip/hip_bf16.h>

// ---------------------------------------------------------------------------
// CDNA5 (gfx1250, wave32) implementation of the windowed-transformer layer.
// GEMMs: 64x64 block tiles, async global->LDS double buffering, WMMA f16.
// Attention: flash-style, dh=32 == WMMA K.
// ---------------------------------------------------------------------------

typedef __attribute__((ext_vector_type(16))) _Float16 v16h;
typedef __attribute__((ext_vector_type(8)))  float    v8f;

#define EMBED 256
#define NHEAD 8
#define DH    32
#define DFF   1024
#define BB    4
#define HH    128
#define WW2   128
#define WSZ   16
#define NW    64
#define TT    256
#define MROWS 65536   // B*NW*T == B*T*NW (flattened GEMM rows for both layers)

union FragA { v16h v; unsigned u[8]; _Float16 h[16]; };
union FragC { v8f  v; float f[8]; };

// Pair-start K index for 16-bit A/B fragments (ISA 7.12.2, 16-bit A 16x32):
// lanes 0-15: VGPR0-3 -> K=0..7, VGPR4-7 -> K=16..23; lanes 16-31: +8.
__device__ __forceinline__ int frag_k(int p, int hi) {
  return ((p < 4) ? (2 * p) : (16 + 2 * (p - 4))) + 8 * hi;
}

// ---------------------------------------------------------------------------
// Tiled WMMA GEMM:  Y[M,N] = X[M,K](f16) @ W[N,K]^T(f16) + bias (+relu)
// Block = 128 threads = 4 waves; block tile 64x64; K-step 32.
// Each wave owns a 32x32 quadrant (2x2 WMMA accumulators).
// Global->LDS staging uses async-LDS loads (ASYNCcnt), double buffered:
// tile k+1 is in flight while tile k feeds the WMMAs.
// ---------------------------------------------------------------------------
__global__ __launch_bounds__(128)
void gemm_xwt(const _Float16* __restrict__ X, const _Float16* __restrict__ Wt,
              const float* __restrict__ bias, float* __restrict__ Yf,
              _Float16* __restrict__ Yh, int M, int N, int K, int relu)
{
  __shared__ _Float16 As[2][64][40];   // 80B rows -> 16B-aligned segments
  __shared__ _Float16 Bs[2][64][40];
  const int tid = threadIdx.x, lane = tid & 31, wv = tid >> 5;
  const int m0 = blockIdx.x * 64, n0 = blockIdx.y * 64;
  const int wm = (wv >> 1) * 32, wn = (wv & 1) * 32;   // wave quadrant
  const int mrow = lane & 15, hi = lane >> 4;
  // cooperative staging: 64 rows x 32 halfs per tile; thread: row tid>>1,
  // 16-half segment (tid&1)*16 -> 32 bytes = 2 async b128 ops per matrix.
  const int lr = tid >> 1;
  const int lc = (tid & 1) * 16;

  const _Float16* pa = X  + (size_t)(m0 + lr) * K + lc;
  const _Float16* pb = Wt + (size_t)(n0 + lr) * K + lc;

  auto issue_tile = [&](int buf, int k0) {
    // LDS aperture keeps the allocation-relative offset in addr[31:0].
    const unsigned la = (unsigned)(size_t)&As[buf][lr][lc];
    const unsigned lb = (unsigned)(size_t)&Bs[buf][lr][lc];
    const unsigned long long ga = (unsigned long long)(size_t)(pa + k0);
    const unsigned long long gb = (unsigned long long)(size_t)(pb + k0);
    // INST_OFFSET applies to both the global and the LDS address (ISA 10.x).
    asm volatile("global_load_async_to_lds_b128 %0, %1, off"
                 :: "v"(la), "v"(ga) : "memory");
    asm volatile("global_load_async_to_lds_b128 %0, %1, off offset:16"
                 :: "v"(la), "v"(ga) : "memory");
    asm volatile("global_load_async_to_lds_b128 %0, %1, off"
                 :: "v"(lb), "v"(gb) : "memory");
    asm volatile("global_load_async_to_lds_b128 %0, %1, off offset:16"
                 :: "v"(lb), "v"(gb) : "memory");
  };

  FragC acc[2][2];
  #pragma unroll
  for (int mi = 0; mi < 2; ++mi)
    #pragma unroll
    for (int ni = 0; ni < 2; ++ni)
      #pragma unroll
      for (int r = 0; r < 8; ++r) acc[mi][ni].f[r] = 0.f;

  const int nk = K >> 5;
  issue_tile(0, 0);
  for (int kt = 0; kt < nk; ++kt) {
    const int buf = kt & 1;
    if (kt + 1 < nk) {
      issue_tile(buf ^ 1, (kt + 1) << 5);
      if (kt + 2 < nk) {
        __builtin_prefetch(pa + ((kt + 2) << 5), 0, 0);
        __builtin_prefetch(pb + ((kt + 2) << 5), 0, 0);
      }
      // async ops complete in order: <=4 outstanding means tile kt is resident
      asm volatile("s_wait_asynccnt 0x4" ::: "memory");
    } else {
      asm volatile("s_wait_asynccnt 0x0" ::: "memory");
    }
    __syncthreads();
    FragA a[2], b[2];
    #pragma unroll
    for (int i = 0; i < 2; ++i) {
      #pragma unroll
      for (int p = 0; p < 8; ++p) {
        const int kb = frag_k(p, hi);
        a[i].u[p] = *(const unsigned*)&As[buf][wm + i * 16 + mrow][kb];
        b[i].u[p] = *(const unsigned*)&Bs[buf][wn + i * 16 + mrow][kb];
      }
    }
    #pragma unroll
    for (int mi = 0; mi < 2; ++mi)
      #pragma unroll
      for (int ni = 0; ni < 2; ++ni)
        acc[mi][ni].v = __builtin_amdgcn_wmma_f32_16x16x32_f16(
            false, a[mi].v, false, b[ni].v, (short)0, acc[mi][ni].v, false, false);
    __syncthreads();
  }

  #pragma unroll
  for (int ni = 0; ni < 2; ++ni) {
    const int col = n0 + wn + ni * 16 + mrow;
    const float bv = bias ? bias[col] : 0.f;
    #pragma unroll
    for (int mi = 0; mi < 2; ++mi) {
      #pragma unroll
      for (int r = 0; r < 8; ++r) {
        const int row = m0 + wm + mi * 16 + r + hi * 8;
        float v = acc[mi][ni].f[r] + bv;
        if (relu) v = fmaxf(v, 0.f);
        if (Yf) Yf[(size_t)row * N + col] = v;
        if (Yh) Yh[(size_t)row * N + col] = (_Float16)v;
      }
    }
  }
}

// ---------------------------------------------------------------------------
// Flash attention, one block per (sequence, head), 256 threads = 8 waves.
// dh = 32 == WMMA K.  Each wave owns 2x16 query rows; K/V streamed through
// LDS in 32-key tiles; online softmax in the C-fragment row mapping.
// ---------------------------------------------------------------------------
__global__ __launch_bounds__(256)
void attn_flash(const _Float16* __restrict__ Qh, const _Float16* __restrict__ Kh,
                const _Float16* __restrict__ Vh, _Float16* __restrict__ AO,
                int L, float scale, const float* __restrict__ maskB, int seq_per_b)
{
  __shared__ _Float16 Kt[32][40];       // [key][dh]
  __shared__ _Float16 Vt[32][40];       // [dh][key]  (transposed on load)
  __shared__ _Float16 Pst[8][16][40];   // per-wave P staging (16 x 32)

  const int seq = blockIdx.x, h = blockIdx.y;
  const int tid = threadIdx.x, lane = tid & 31, wv = tid >> 5;
  const int mrow = lane & 15, hi = lane >> 4;
  const int q0 = wv * 32;
  const bool active = q0 < L;

  const float* mr = maskB ? (maskB + (size_t)(seq / seq_per_b) * L) : nullptr;

  FragA qf[2];
  FragC o0[2], o1[2];
  float om[2][8], ol[2][8];
  if (active) {
    #pragma unroll
    for (int qt = 0; qt < 2; ++qt) {
      const _Float16* qb = Qh + ((size_t)seq * L + q0 + qt * 16 + mrow) * EMBED + h * DH;
      #pragma unroll
      for (int p = 0; p < 8; ++p) qf[qt].u[p] = *(const unsigned*)(qb + frag_k(p, hi));
      #pragma unroll
      for (int r = 0; r < 8; ++r) {
        om[qt][r] = -1e30f; ol[qt][r] = 0.f; o0[qt].f[r] = 0.f; o1[qt].f[r] = 0.f;
      }
    }
  }

  const int nkt = L >> 5;
  for (int j = 0; j < nkt; ++j) {
    { // cooperative K/V tile loads (V transposed so PV B-frags are contiguous)
      const int row = tid >> 3, seg = (tid & 7) * 4;
      const size_t base = ((size_t)seq * L + j * 32 + row) * EMBED + h * DH + seg;
      *(float2*)&Kt[row][seg] = *(const float2*)(Kh + base);
      _Float16 v4[4];
      *(float2*)v4 = *(const float2*)(Vh + base);
      #pragma unroll
      for (int q = 0; q < 4; ++q) Vt[seg + q][row] = v4[q];
    }
    __syncthreads();
    if (active) {
      #pragma unroll
      for (int qt = 0; qt < 2; ++qt) {
        FragA kf0, kf1;
        #pragma unroll
        for (int p = 0; p < 8; ++p) {
          const int kb = frag_k(p, hi);
          kf0.u[p] = *(const unsigned*)&Kt[mrow][kb];       // keys 0..15
          kf1.u[p] = *(const unsigned*)&Kt[16 + mrow][kb];  // keys 16..31
        }
        v8f zf = {};
        FragC s0, s1;
        s0.v = __builtin_amdgcn_wmma_f32_16x16x32_f16(false, qf[qt].v, false, kf0.v, (short)0, zf, false, false);
        s1.v = __builtin_amdgcn_wmma_f32_16x16x32_f16(false, qf[qt].v, false, kf1.v, (short)0, zf, false, false);
        #pragma unroll
        for (int r = 0; r < 8; ++r) {
          float a = s0.f[r] * scale, b = s1.f[r] * scale;
          if (mr) { a += mr[j * 32 + mrow]; b += mr[j * 32 + 16 + mrow]; }
          float m = fmaxf(a, b);
          #pragma unroll
          for (int off = 8; off; off >>= 1) m = fmaxf(m, __shfl_xor(m, off, 16));
          const float mnew = fmaxf(om[qt][r], m);
          const float corr = __expf(om[qt][r] - mnew);
          const float p0 = __expf(a - mnew);
          const float p1 = __expf(b - mnew);
          float rs = p0 + p1;
          #pragma unroll
          for (int off = 8; off; off >>= 1) rs += __shfl_xor(rs, off, 16);
          ol[qt][r] = ol[qt][r] * corr + rs;
          om[qt][r] = mnew;
          o0[qt].f[r] *= corr;
          o1[qt].f[r] *= corr;
          Pst[wv][r + hi * 8][mrow]      = (_Float16)p0;
          Pst[wv][r + hi * 8][16 + mrow] = (_Float16)p1;
        }
        // wave-local LDS RAW: P staging -> A-fragment readback
        asm volatile("s_wait_dscnt 0x0" ::: "memory");
        FragA pf, vf0, vf1;
        #pragma unroll
        for (int p = 0; p < 8; ++p) {
          const int kb = frag_k(p, hi);
          pf.u[p]  = *(const unsigned*)&Pst[wv][mrow][kb];
          vf0.u[p] = *(const unsigned*)&Vt[mrow][kb];       // dh 0..15
          vf1.u[p] = *(const unsigned*)&Vt[16 + mrow][kb];  // dh 16..31
        }
        o0[qt].v = __builtin_amdgcn_wmma_f32_16x16x32_f16(false, pf.v, false, vf0.v, (short)0, o0[qt].v, false, false);
        o1[qt].v = __builtin_amdgcn_wmma_f32_16x16x32_f16(false, pf.v, false, vf1.v, (short)0, o1[qt].v, false, false);
      }
    }
    __syncthreads();
  }

  if (active) {
    #pragma unroll
    for (int qt = 0; qt < 2; ++qt) {
      #pragma unroll
      for (int r = 0; r < 8; ++r) {
        const float inv = 1.f / ol[qt][r];
        const int row = q0 + qt * 16 + r + hi * 8;
        _Float16* dst = AO + ((size_t)seq * L + row) * EMBED + h * DH;
        dst[mrow]      = (_Float16)(o0[qt].f[r] * inv);
        dst[16 + mrow] = (_Float16)(o1[qt].f[r] * inv);
      }
    }
  }
}

// ---------------------------------------------------------------------------
// Residual add + LayerNorm over C=256 (one block per row).
// ---------------------------------------------------------------------------
__global__ __launch_bounds__(256)
void add_layernorm(const float* __restrict__ X, const float* __restrict__ Y,
                   const float* __restrict__ g, const float* __restrict__ b,
                   float* __restrict__ Of, _Float16* __restrict__ Oh)
{
  __shared__ float red[8];
  const int row = blockIdx.x, c = threadIdx.x;
  const size_t idx = (size_t)row * EMBED + c;
  const float v = X[idx] + Y[idx];
  float s = v;
  #pragma unroll
  for (int off = 16; off; off >>= 1) s += __shfl_xor(s, off, 32);
  if ((c & 31) == 0) red[c >> 5] = s;
  __syncthreads();
  float tot = 0.f;
  #pragma unroll
  for (int i = 0; i < 8; ++i) tot += red[i];
  const float mean = tot * (1.f / EMBED);
  const float d = v - mean;
  float s2 = d * d;
  __syncthreads();
  #pragma unroll
  for (int off = 16; off; off >>= 1) s2 += __shfl_xor(s2, off, 32);
  if ((c & 31) == 0) red[c >> 5] = s2;
  __syncthreads();
  float tot2 = 0.f;
  #pragma unroll
  for (int i = 0; i < 8; ++i) tot2 += red[i];
  const float rstd = rsqrtf(tot2 * (1.f / EMBED) + 1e-5f);
  const float o = d * rstd * g[c] + b[c];
  Of[idx] = o;
  if (Oh) Oh[idx] = (_Float16)o;
}

// ---------------------------------------------------------------------------
// Small helpers / data movement
// ---------------------------------------------------------------------------
__global__ void f2h(const float* __restrict__ s, _Float16* __restrict__ d, int n) {
  const int i = blockIdx.x * 256 + threadIdx.x;
  if (i < n) d[i] = (_Float16)s[i];
}

// rel[t][c] = relu(coords @ w1^T + b1) @ w2^T + b2
__global__ __launch_bounds__(256)
void rel_kernel(const float* __restrict__ w1, const float* __restrict__ b1,
                const float* __restrict__ w2, const float* __restrict__ b2,
                float* __restrict__ rel)
{
  __shared__ float hid[64];
  const int t = blockIdx.x, c = threadIdx.x;
  const float y = (float)(t >> 4) * (1.f / 15.f);
  const float x = (float)(t & 15) * (1.f / 15.f);
  if (c < 64) hid[c] = fmaxf(0.f, w1[c * 2] * y + w1[c * 2 + 1] * x + b1[c]);
  __syncthreads();
  float acc = b2[c];
  #pragma unroll 8
  for (int j = 0; j < 64; ++j) acc += hid[j] * w2[c * 64 + j];
  rel[t * EMBED + c] = acc;
}

// POSW[w][t][c] = glob_pos[c][h][x] + rel[t][c]  (fp16)
__global__ __launch_bounds__(256)
void pos_kernel(const float* __restrict__ glob, const float* __restrict__ rel,
                _Float16* __restrict__ POSW)
{
  const int wt = blockIdx.x, c = threadIdx.x;
  const int w = wt >> 8, t = wt & 255;
  const int hh = (w >> 3) * WSZ + (t >> 4);
  const int xx = (w & 7) * WSZ + (t & 15);
  const float v = glob[((size_t)c * HH + hh) * WW2 + xx] + rel[t * EMBED + c];
  POSW[(size_t)wt * EMBED + c] = (_Float16)v;
}

// valid[b][w] = maxpool(defe window) > 0 ; maskB = 0 / -1e30
__global__ __launch_bounds__(256)
void valid_kernel(const float* __restrict__ defe, float* __restrict__ valid,
                  float* __restrict__ maskB)
{
  __shared__ float red[8];
  const int bw = blockIdx.x, t = threadIdx.x;
  const int b = bw >> 6, w = bw & 63;
  const int hh = (w >> 3) * WSZ + (t >> 4);
  const int xx = (w & 7) * WSZ + (t & 15);
  float v = defe[((size_t)b * HH + hh) * WW2 + xx];
  #pragma unroll
  for (int off = 16; off; off >>= 1) v = fmaxf(v, __shfl_xor(v, off, 32));
  if ((t & 31) == 0) red[t >> 5] = v;
  __syncthreads();
  if (t == 0) {
    float m = red[0];
    for (int i = 1; i < 8; ++i) m = fmaxf(m, red[i]);
    const bool ok = m > 0.f;
    valid[bw] = ok ? 1.f : 0.f;
    maskB[bw] = ok ? 0.f : -1e30f;
  }
}

// Window partition: XW (f32 residual), QK16 = x+pos (f16), V16 = x (f16)
__global__ __launch_bounds__(256)
void build_win(const float* __restrict__ backbone, const _Float16* __restrict__ POSW,
               float* __restrict__ XW, _Float16* __restrict__ QK16,
               _Float16* __restrict__ V16)
{
  const int r = blockIdx.x;          // (b*NW + w)*T + t
  const int c = threadIdx.x;
  const int b = r >> 14;
  const int w = (r >> 8) & 63;
  const int t = r & 255;
  const int hh = (w >> 3) * WSZ + (t >> 4);
  const int xx = (w & 7) * WSZ + (t & 15);
  const float xv = backbone[(((size_t)b * EMBED + c) * HH + hh) * WW2 + xx];
  const float pv = (float)POSW[((size_t)w * TT + t) * EMBED + c];
  const size_t idx = (size_t)r * EMBED + c;
  XW[idx] = xv;
  QK16[idx] = (_Float16)(xv + pv);
  V16[idx] = (_Float16)xv;
}

// Transpose to cross-window layout [(b*T + t)*NW + w]
__global__ __launch_bounds__(256)
void build_cross(const float* __restrict__ XW, const _Float16* __restrict__ POSW,
                 float* __restrict__ XC, _Float16* __restrict__ QK16,
                 _Float16* __restrict__ V16)
{
  const int r2 = blockIdx.x;         // (b*T + t)*NW + w
  const int c = threadIdx.x;
  const int b = r2 >> 14;
  const int t = (r2 >> 6) & 255;
  const int w = r2 & 63;
  const size_t r1 = ((size_t)b * NW + w) * TT + t;
  const float xv = XW[r1 * EMBED + c];
  const float pv = (float)POSW[((size_t)w * TT + t) * EMBED + c];
  const size_t idx = (size_t)r2 * EMBED + c;
  XC[idx] = xv;
  QK16[idx] = (_Float16)(xv + pv);
  V16[idx] = (_Float16)xv;
}

// out = backbone + valid * enc (scatter cross-layout back to [B,C,H,W])
__global__ __launch_bounds__(256)
void final_kernel(const float* __restrict__ backbone, const float* __restrict__ XC,
                  const float* __restrict__ valid, float* __restrict__ out)
{
  const size_t i = (size_t)blockIdx.x * 256 + threadIdx.x;
  const int xx = (int)(i & 127);
  const int hh = (int)((i >> 7) & 127);
  const int c  = (int)((i >> 14) & 255);
  const int b  = (int)(i >> 22);
  const int w  = (hh >> 4) * 8 + (xx >> 4);
  const int t  = (hh & 15) * 16 + (xx & 15);
  const size_t r2 = ((size_t)b * TT + t) * NW + w;
  out[i] = backbone[i] + XC[r2 * EMBED + c] * valid[b * NW + w];
}

__global__ void copy_valid(const float* __restrict__ v, float* __restrict__ out) {
  out[threadIdx.x] = v[threadIdx.x];
}

// ---------------------------------------------------------------------------
extern "C" void kernel_launch(void* const* d_in, const int* in_sizes, int n_in,
                              void* d_out, int out_size, void* d_ws, size_t ws_size,
                              hipStream_t stream)
{
  const float* backbone = (const float*)d_in[0];
  const float* defe     = (const float*)d_in[1];
  // d_in[2] = window_size (fixed 16, compiled in)
  const float* glob     = (const float*)d_in[3];
  const float* rel_w1   = (const float*)d_in[4];
  const float* rel_b1   = (const float*)d_in[5];
  const float* rel_w2   = (const float*)d_in[6];
  const float* rel_b2   = (const float*)d_in[7];
  const float* in_w     = (const float*)d_in[8];
  const float* in_b     = (const float*)d_in[9];
  const float* out_w    = (const float*)d_in[10];
  const float* out_b    = (const float*)d_in[11];
  const float* l1w      = (const float*)d_in[12];
  const float* l1b      = (const float*)d_in[13];
  const float* l2w      = (const float*)d_in[14];
  const float* l2b      = (const float*)d_in[15];
  const float* g1       = (const float*)d_in[16];
  const float* b1       = (const float*)d_in[17];
  const float* g2       = (const float*)d_in[18];
  const float* b2       = (const float*)d_in[19];
  (void)in_sizes; (void)n_in; (void)out_size; (void)ws_size;

  char* ws = (char*)d_ws;
  size_t o = 0;
  auto alloc = [&](size_t bytes) { size_t r = o; o += (bytes + 255) & ~(size_t)255; return r; };
  const size_t oXW   = alloc((size_t)MROWS * EMBED * 4);   // residual, window layout
  const size_t oXC   = alloc((size_t)MROWS * EMBED * 4);   // residual, cross layout
  const size_t oSRC  = alloc((size_t)MROWS * EMBED * 4);   // post-LN1
  const size_t oPF   = alloc((size_t)MROWS * EMBED * 4);   // f32 GEMM outputs
  const size_t oPOS  = alloc((size_t)NW * TT * EMBED * 2);
  const size_t oQK   = alloc((size_t)MROWS * EMBED * 2);
  const size_t oV16  = alloc((size_t)MROWS * EMBED * 2);
  const size_t oQH   = alloc((size_t)MROWS * EMBED * 2);   // FF1H overlays QH..AO
  const size_t oKH   = alloc((size_t)MROWS * EMBED * 2);
  const size_t oVH   = alloc((size_t)MROWS * EMBED * 2);
  const size_t oAO   = alloc((size_t)MROWS * EMBED * 2);
  const size_t oREL  = alloc((size_t)TT * EMBED * 4);
  const size_t oW16  = alloc((size_t)(3 * EMBED * EMBED + EMBED * EMBED + DFF * EMBED + EMBED * DFF) * 2);
  const size_t oVAL  = alloc(BB * NW * 4);
  const size_t oMSK  = alloc(BB * NW * 4);

  float*    XW   = (float*)(ws + oXW);
  float*    XC   = (float*)(ws + oXC);
  float*    SRC  = (float*)(ws + oSRC);
  float*    PF   = (float*)(ws + oPF);
  _Float16* POSW = (_Float16*)(ws + oPOS);
  _Float16* QK16 = (_Float16*)(ws + oQK);
  _Float16* V16  = (_Float16*)(ws + oV16);
  _Float16* QH   = (_Float16*)(ws + oQH);
  _Float16* KH   = (_Float16*)(ws + oKH);
  _Float16* VH   = (_Float16*)(ws + oVH);
  _Float16* AO   = (_Float16*)(ws + oAO);
  _Float16* FF1H = QH;   // 128MB overlay of QH/KH/VH/AO (dead by FF1 time)
  float*    REL  = (float*)(ws + oREL);
  _Float16* W16  = (_Float16*)(ws + oW16);
  float*    VALID= (float*)(ws + oVAL);
  float*    MASK = (float*)(ws + oMSK);

  _Float16* Wq = W16;                          // in_proj (3C x C)
  _Float16* Wo = W16 + 3 * EMBED * EMBED;      // out_proj (C x C)
  _Float16* W1 = Wo + EMBED * EMBED;           // lin1 (DFF x C)
  _Float16* W2 = W1 + DFF * EMBED;             // lin2 (C x DFF)

  // Weight fp32 -> fp16 (once per launch; tiny)
  f2h<<<(3 * EMBED * EMBED + 255) / 256, 256, 0, stream>>>(in_w, Wq, 3 * EMBED * EMBED);
  f2h<<<(EMBED * EMBED + 255) / 256, 256, 0, stream>>>(out_w, Wo, EMBED * EMBED);
  f2h<<<(DFF * EMBED + 255) / 256, 256, 0, stream>>>(l1w, W1, DFF * EMBED);
  f2h<<<(EMBED * DFF + 255) / 256, 256, 0, stream>>>(l2w, W2, EMBED * DFF);

  rel_kernel<<<TT, 256, 0, stream>>>(rel_w1, rel_b1, rel_w2, rel_b2, REL);
  pos_kernel<<<NW * TT, 256, 0, stream>>>(glob, REL, POSW);
  valid_kernel<<<BB * NW, 256, 0, stream>>>(defe, VALID, MASK);
  build_win<<<MROWS, 256, 0, stream>>>(backbone, POSW, XW, QK16, V16);

  const float scale = 0.17677669529663687f;  // 1/sqrt(DH)
  const dim3 gB(128);
  const dim3 gC(MROWS / 64, EMBED / 64);
  const dim3 gF(MROWS / 64, DFF / 64);

  auto enc = [&](float* Xres, int L, int S, const float* maskB) {
    gemm_xwt<<<gC, gB, 0, stream>>>(QK16, Wq,                     in_b,             nullptr, QH, MROWS, EMBED, EMBED, 0);
    gemm_xwt<<<gC, gB, 0, stream>>>(QK16, Wq + EMBED * EMBED,     in_b + EMBED,     nullptr, KH, MROWS, EMBED, EMBED, 0);
    gemm_xwt<<<gC, gB, 0, stream>>>(V16,  Wq + 2 * EMBED * EMBED, in_b + 2 * EMBED, nullptr, VH, MROWS, EMBED, EMBED, 0);
    attn_flash<<<dim3(S, NHEAD), 256, 0, stream>>>(QH, KH, VH, AO, L, scale, maskB, TT);
    gemm_xwt<<<gC, gB, 0, stream>>>(AO, Wo, out_b, PF, nullptr, MROWS, EMBED, EMBED, 0);
    add_layernorm<<<MROWS, 256, 0, stream>>>(Xres, PF, g1, b1, SRC, V16);  // V16 reused as srcH
    gemm_xwt<<<gF, gB, 0, stream>>>(V16, W1, l1b, nullptr, FF1H, MROWS, DFF, EMBED, 1);
    gemm_xwt<<<gC, gB, 0, stream>>>(FF1H, W2, l2b, PF, nullptr, MROWS, EMBED, DFF, 0);
    add_layernorm<<<MROWS, 256, 0, stream>>>(SRC, PF, g2, b2, Xres, nullptr);
  };

  // within-window attention: 256 sequences of L=256, no mask
  enc(XW, TT, BB * NW, nullptr);

  // cross-window attention: 1024 sequences of L=64, key-padding mask per batch
  build_cross<<<MROWS, 256, 0, stream>>>(XW, POSW, XC, QK16, V16);
  enc(XC, NW, BB * TT, MASK);

  final_kernel<<<(BB * EMBED * HH * WW2) / 256, 256, 0, stream>>>(backbone, XC, VALID, (float*)d_out);
  copy_valid<<<1, 256, 0, stream>>>(VALID, (float*)d_out + (size_t)BB * EMBED * HH * WW2);
}
